// HeteroGNN1_76802605187592
// MI455X (gfx1250) — compile-verified, matrix operand order checked by reading, not run on hardware
//
#include <hip/hip_runtime.h>

// ---------- vector types ----------
typedef _Float16 v16h __attribute__((ext_vector_type(16)));
typedef _Float16 h8   __attribute__((ext_vector_type(8)));
typedef _Float16 h4   __attribute__((ext_vector_type(4)));
typedef float    v8f  __attribute__((ext_vector_type(8)));
typedef float    f4v  __attribute__((ext_vector_type(4)));

// Build a 16x16x32 f16 WMMA operand fragment from a row-contiguous array of K
// halves (LDS A-tile row, or a row of the transposed weight matrix WT[n][*]).
// Layout per CDNA5 ISA 7.12.2 (16-bit A 16x32): lane group lo/hi selects K+0/K+8;
// halves 0..7 -> K = kb + hi*8 + e ; halves 8..15 -> K = kb + 16 + hi*8 + (e-8).
__device__ __forceinline__ v16h frag16(const _Float16* __restrict__ row, int kb, int hi) {
  h8 lo = *(const h8*)(row + kb + hi * 8);
  h8 hh = *(const h8*)(row + kb + 16 + hi * 8);
  v16h a;
#pragma unroll
  for (int i = 0; i < 8; ++i) { a[i] = lo[i]; a[i + 8] = hh[i]; }
  return a;
}

// ---------- weight convert: W[K][N] f32 -> WT[N][K] f16 ----------
__global__ void wt_f16_kernel(const float* __restrict__ W, _Float16* __restrict__ WT,
                              int K, int Nout) {
  int i = blockIdx.x * blockDim.x + threadIdx.x;
  if (i < K * Nout) {
    int k = i / Nout, n = i % Nout;
    WT[(size_t)n * K + k] = (_Float16)W[i];
  }
}

// ---------- degree count ----------
__global__ void count_kernel(const int* __restrict__ dst, float* __restrict__ cnt, int E) {
  int e = blockIdx.x * blockDim.x + threadIdx.x;
  if (e < E) atomicAdd(&cnt[dst[e]], 1.0f);
}

// ---------- edge scatter-add: agg[dst] += feat[src] (f32 accumulate) ----------
template <typename T, int F>
__global__ void scatter_kernel(const int* __restrict__ src, const int* __restrict__ dst,
                               const T* __restrict__ feat, float* __restrict__ agg, int E) {
  const int TPE = F / 4;  // threads per edge, 4 features each
  long long gid = (long long)blockIdx.x * blockDim.x + threadIdx.x;
  if (gid >= (long long)E * TPE) return;
  int e  = (int)(gid / TPE);
  int fq = (int)(gid % TPE);
  int s = src[e], d = dst[e];
  float v[4];
  if constexpr (sizeof(T) == 4) {
    f4v t = ((const f4v*)(feat + (size_t)s * F))[fq];
    v[0] = t[0]; v[1] = t[1]; v[2] = t[2]; v[3] = t[3];
  } else {
    h4 t = ((const h4*)(feat + (size_t)s * F))[fq];
    v[0] = (float)t[0]; v[1] = (float)t[1]; v[2] = (float)t[2]; v[3] = (float)t[3];
  }
  float* a = agg + (size_t)d * F + fq * 4;
#pragma unroll
  for (int j = 0; j < 4; ++j) atomicAdd(a + j, v[j]);
}

// ---------- coalesced epilogue: D tile (LDS) -> global, 16B per thread ----------
template <typename TOUT, int HO>
__device__ __forceinline__ void store_tile(const TOUT (*Ds)[HO + 8],
                                           TOUT* __restrict__ out,
                                           int nb, int nNodes, int nThreads) {
  typedef TOUT vstore __attribute__((ext_vector_type(16 / sizeof(TOUT))));
  constexpr int VE = 16 / sizeof(TOUT);
  for (int idx = threadIdx.x * VE; idx < 16 * HO; idx += nThreads * VE) {
    int r = idx / HO, c0 = idx % HO;
    if (nb + r < nNodes)
      *(vstore*)(out + (size_t)(nb + r) * HO + c0) = *(const vstore*)&Ds[r][c0];
  }
}

// ---------- fused SAGE GEMM: out = relu( (agg/cnt)@Wl + x2@Wr + b ), f16 out ----------
// Block: 16 node rows x 128 output cols; 8 waves, one 16-col N-tile per wave.
template <typename TA2, int K>
__global__ __launch_bounds__(256) void sage_gemm_kernel(
    const float* __restrict__ agg, const float* __restrict__ cnt,
    const TA2* __restrict__ x2,
    const _Float16* __restrict__ WlT, const _Float16* __restrict__ WrT,
    const float* __restrict__ bias, _Float16* __restrict__ out, int nNodes) {
  constexpr int PK = K + 8;  // pad rows to break LDS bank conflicts
  __shared__ __align__(16) _Float16 As1[16][PK];
  __shared__ __align__(16) _Float16 As2[16][PK];
  __shared__ __align__(16) _Float16 Ds[16][128 + 8];
  __shared__ float invs[16];
  const int nb = blockIdx.x * 16;
  if (threadIdx.x < 16) {
    int node = nb + threadIdx.x; if (node >= nNodes) node = nNodes - 1;
    invs[threadIdx.x] = 1.0f / fmaxf(cnt[node], 1.0f);
  }
  __syncthreads();
  for (int idx = threadIdx.x; idx < 16 * K; idx += blockDim.x) {
    int r = idx / K, c = idx % K;
    int node = nb + r; if (node >= nNodes) node = nNodes - 1;
    As1[r][c] = (_Float16)(agg[(size_t)node * K + c] * invs[r]);
    As2[r][c] = (_Float16)((float)x2[(size_t)node * K + c]);
  }
  __syncthreads();

  const int wave = threadIdx.x >> 5;        // N-tile
  const int lane = threadIdx.x & 31;
  const int r    = lane & 15;
  const int hi   = lane >> 4;
  const int ncol = wave * 16 + r;
  v8f c = {};
#pragma unroll
  for (int kb = 0; kb < K; kb += 32) {
    v16h a = frag16(&As1[r][0], kb, hi);
    v16h b = frag16(&WlT[(size_t)ncol * K], kb, hi);
    c = __builtin_amdgcn_wmma_f32_16x16x32_f16(false, a, false, b, (short)0, c, false, false);
  }
#pragma unroll
  for (int kb = 0; kb < K; kb += 32) {
    v16h a = frag16(&As2[r][0], kb, hi);
    v16h b = frag16(&WrT[(size_t)ncol * K], kb, hi);
    c = __builtin_amdgcn_wmma_f32_16x16x32_f16(false, a, false, b, (short)0, c, false, false);
  }
  float bv = bias[ncol];
#pragma unroll
  for (int i = 0; i < 8; ++i)  // C/D layout: vgpr i -> row i + 8*(lane/16)
    Ds[i + hi * 8][ncol] = (_Float16)fmaxf(c[i] + bv, 0.0f);
  __syncthreads();
  store_tile<_Float16, 128>(Ds, out, nb, nNodes, 256);
}

// ---------- plain linear: out = act(A @ W + b) ----------
template <bool RELU, typename TOUT, int K, int HO, int NTHR>
__global__ __launch_bounds__(256) void lin_gemm_kernel(
    const _Float16* __restrict__ A, const _Float16* __restrict__ WT,
    const float* __restrict__ bias, TOUT* __restrict__ out, int nNodes) {
  constexpr int PK = K + 8;
  __shared__ __align__(16) _Float16 As[16][PK];
  __shared__ __align__(16) TOUT Ds[16][HO + 8];
  const int nb = blockIdx.x * 16;
  for (int idx = threadIdx.x; idx < 16 * K; idx += NTHR) {
    int r = idx / K, cix = idx % K;
    int node = nb + r; if (node >= nNodes) node = nNodes - 1;
    As[r][cix] = A[(size_t)node * K + cix];
  }
  __syncthreads();
  const int wave = threadIdx.x >> 5;
  const int lane = threadIdx.x & 31;
  const int r    = lane & 15;
  const int hi   = lane >> 4;
  const int ncol = wave * 16 + r;
  v8f c = {};
#pragma unroll
  for (int kb = 0; kb < K; kb += 32) {
    v16h a = frag16(&As[r][0], kb, hi);
    v16h b = frag16(&WT[(size_t)ncol * K], kb, hi);
    c = __builtin_amdgcn_wmma_f32_16x16x32_f16(false, a, false, b, (short)0, c, false, false);
  }
  float bv = bias[ncol];
#pragma unroll
  for (int i = 0; i < 8; ++i) {
    float v = c[i] + bv;
    if (RELU) v = fmaxf(v, 0.0f);
    Ds[i + hi * 8][ncol] = (TOUT)v;
  }
  __syncthreads();
  store_tile<TOUT, HO>(Ds, out, nb, nNodes, NTHR);
}

static inline int cdiv(long long a, long long b) { return (int)((a + b - 1) / b); }

extern "C" void kernel_launch(void* const* d_in, const int* in_sizes, int n_in,
                              void* d_out, int out_size, void* d_ws, size_t ws_size,
                              hipStream_t stream) {
  (void)n_in; (void)out_size; (void)ws_size;
  constexpr int D = 64, H = 128, O = 64;
  const float* x     = (const float*)d_in[0];
  const int*   ei    = (const int*)d_in[1];
  const float* W1l   = (const float*)d_in[2];
  const float* b1    = (const float*)d_in[3];
  const float* W1r   = (const float*)d_in[4];
  const float* Wlin1 = (const float*)d_in[5];
  const float* blin1 = (const float*)d_in[6];
  const float* W2l   = (const float*)d_in[7];
  const float* b2    = (const float*)d_in[8];
  const float* W2r   = (const float*)d_in[9];
  const float* Wlin2 = (const float*)d_in[10];
  const float* blin2 = (const float*)d_in[11];

  const int Nn = in_sizes[0] / D;
  const int E  = in_sizes[1] / 2;
  const int* src = ei;
  const int* dst = ei + E;

  // ---- workspace layout ----
  char* p = (char*)d_ws;
  auto alloc = [&](size_t bytes) -> char* {
    char* r = p; p += (bytes + 255) & ~(size_t)255; return r;
  };
  float*     cnt    = (float*)alloc((size_t)Nn * 4);
  float*     agg1   = (float*)alloc((size_t)Nn * D * 4);
  float*     agg2   = (float*)alloc((size_t)Nn * H * 4);
  _Float16*  h1     = (_Float16*)alloc((size_t)Nn * H * 2);
  _Float16*  h1b    = (_Float16*)alloc((size_t)Nn * H * 2);
  _Float16*  h2     = (_Float16*)alloc((size_t)Nn * H * 2);
  _Float16*  W1lT   = (_Float16*)alloc((size_t)D * H * 2);
  _Float16*  W1rT   = (_Float16*)alloc((size_t)D * H * 2);
  _Float16*  Wlin1T = (_Float16*)alloc((size_t)H * H * 2);
  _Float16*  W2lT   = (_Float16*)alloc((size_t)H * H * 2);
  _Float16*  W2rT   = (_Float16*)alloc((size_t)H * H * 2);
  _Float16*  Wlin2T = (_Float16*)alloc((size_t)H * O * 2);

  // ---- zero accumulators (graph-capture safe) ----
  hipMemsetAsync(cnt,  0, (size_t)Nn * 4,     stream);
  hipMemsetAsync(agg1, 0, (size_t)Nn * D * 4, stream);
  hipMemsetAsync(agg2, 0, (size_t)Nn * H * 4, stream);

  // ---- weights -> transposed f16 ----
  wt_f16_kernel<<<cdiv(D * H, 256), 256, 0, stream>>>(W1l,   W1lT,   D, H);
  wt_f16_kernel<<<cdiv(D * H, 256), 256, 0, stream>>>(W1r,   W1rT,   D, H);
  wt_f16_kernel<<<cdiv(H * H, 256), 256, 0, stream>>>(Wlin1, Wlin1T, H, H);
  wt_f16_kernel<<<cdiv(H * H, 256), 256, 0, stream>>>(W2l,   W2lT,   H, H);
  wt_f16_kernel<<<cdiv(H * H, 256), 256, 0, stream>>>(W2r,   W2rT,   H, H);
  wt_f16_kernel<<<cdiv(H * O, 256), 256, 0, stream>>>(Wlin2, Wlin2T, H, O);

  // ---- degree counts ----
  count_kernel<<<cdiv(E, 256), 256, 0, stream>>>(dst, cnt, E);

  const int nblk = cdiv(Nn, 16);

  // ---- layer 1: scatter x -> agg1, fused SAGE GEMM (K=64, f32 A2) ----
  scatter_kernel<float, D><<<cdiv((long long)E * (D / 4), 256), 256, 0, stream>>>(
      src, dst, x, agg1, E);
  sage_gemm_kernel<float, D><<<nblk, 256, 0, stream>>>(
      agg1, cnt, x, W1lT, W1rT, b1, h1, Nn);

  // ---- hidden linear: h1b = relu(h1 @ Wlin1 + blin1) ----
  lin_gemm_kernel<true, _Float16, H, H, 256><<<nblk, 256, 0, stream>>>(
      h1, Wlin1T, blin1, h1b, Nn);

  // ---- layer 2: scatter h1b -> agg2, fused SAGE GEMM (K=128, f16 A2) ----
  scatter_kernel<_Float16, H><<<cdiv((long long)E * (H / 4), 256), 256, 0, stream>>>(
      src, dst, h1b, agg2, E);
  sage_gemm_kernel<_Float16, H><<<nblk, 256, 0, stream>>>(
      agg2, cnt, h1b, W2lT, W2rT, b2, h2, Nn);

  // ---- output linear: d_out = h2 @ Wlin2 + blin2 (f32, no relu, HO=64 -> 4 waves) ----
  lin_gemm_kernel<false, float, H, O, 128><<<nblk, 128, 0, stream>>>(
      h2, Wlin2T, blin2, (float*)d_out, Nn);
}
